// TotalLoss_74268574482788
// MI455X (gfx1250) — matrix operand order repaired and checked by reading, not compiled, not run on hardware
//
#include <hip/hip_runtime.h>
#include <hip/hip_bf16.h>

typedef __attribute__((ext_vector_type(16))) _Float16 v16h;
typedef __attribute__((ext_vector_type(8)))  _Float16 v8h;
typedef __attribute__((ext_vector_type(8)))  float    v8f;

#define NB   8
#define NH   256
#define NW   256
#define NPIX (NB * NH * NW)            // 524288
#define FEATN (8 * 512 * 32 * 32)      // 4194304 final VGG feature elements
#define THRESH (70.0f / 127.0f)

// ---------------------------------------------------------------------------
// zero the scalar accumulators (ws[0..15] floats) each launch
// ---------------------------------------------------------------------------
__global__ void zero_acc_kernel(float* acc) {
    if (threadIdx.x < 16) acc[threadIdx.x] = 0.0f;
}

// ---------------------------------------------------------------------------
// pack f32 NCHW input into f16 NHWC [B,H,W,3]; nchan_src==1 replicates gray
// ---------------------------------------------------------------------------
__global__ void pack_input_kernel(const float* __restrict__ src,
                                  _Float16* __restrict__ dst, int nchan_src) {
    size_t i = (size_t)blockIdx.x * blockDim.x + threadIdx.x;
    size_t HW = (size_t)NH * NW;
    size_t total = (size_t)NB * HW * 3;
    if (i >= total) return;
    int c = (int)(i % 3);
    size_t t = i / 3;
    size_t w = t % NW; t /= NW;
    size_t h = t % NH;
    size_t b = t / NH;
    float v = (nchan_src == 1) ? src[(b * NH + h) * NW + w]
                               : src[((b * 3 + c) * NH + h) * NW + w];
    dst[i] = (_Float16)v;
}

// ---------------------------------------------------------------------------
// pack layer weights: f32 OIHW -> f16 [Cout][kpad] row-major, K zero-padded.
// GEMM K-ordering: k = (kh*3+kw)*Cin + ci  (tap-major, channels contiguous).
// ---------------------------------------------------------------------------
__global__ void pack_weights_kernel(const float* __restrict__ W,
                                    _Float16* __restrict__ Wp,
                                    int Cout, int Cin, int kpad) {
    int Ktot = Cin * 9;
    size_t i = (size_t)blockIdx.x * blockDim.x + threadIdx.x;
    size_t total = (size_t)Cout * kpad;
    if (i >= total) return;
    int k = (int)(i % kpad);
    int m = (int)(i / kpad);
    _Float16 v = (_Float16)0.0f;
    if (k < Ktot) {
        int r  = k / Cin;          // filter tap kh*3+kw
        int ci = k - r * Cin;
        v = (_Float16)W[((size_t)m * Cin + ci) * 9 + r];
    }
    Wp[i] = v;
}

// ---------------------------------------------------------------------------
// Implicit-GEMM 3x3 conv (pad 1, stride 1), NHWC f16, WMMA f32_16x16x32_f16.
// Block = 128 threads (4 waves), tile = 64 Cout x 64 spatial positions.
// K-slice of 32 = one filter tap x 32 contiguous channels (when Cin%32==0):
// staging is two global_load_async_to_lds_b128 per thread (ASYNCcnt-tracked,
// overlapped behind the WMMAs via LDS double-buffering; one s_wait_asynccnt
// + barrier per K-step). Zero-padding lanes do a plain LDS vector store.
// ---------------------------------------------------------------------------
__global__ __launch_bounds__(128)
void conv_wmma_kernel(const _Float16* __restrict__ X,
                      const _Float16* __restrict__ Wp,
                      const float* __restrict__ Bias,
                      _Float16* __restrict__ Y,
                      int Cin, int Cout, int H, int Wd, int relu, int kpad) {
    __shared__ __align__(32) _Float16 ldsB[2][64 * 32];   // [buf][n][k], 8KB

    const int tid  = threadIdx.x;
    const int lane = tid & 31;
    const int wave = tid >> 5;
    const int Ktot = Cin * 9;
    const int HW   = H * Wd;
    const int n0   = blockIdx.x * 64;

    // ---- WMMA fragment indices (contiguous k = 16*(lane/16) + s in A and B) -----------------------
    const int bN     = lane & 15;
    const int kbaseA = (lane < 16) ? 0 : 16;
    const int kbaseB = (lane < 16) ? 0 : 16;
    const int m      = ((blockIdx.y * 4 + wave) << 4) + (lane & 15);

    // ---- staging role: thread covers n = n0 + nn, k-halves khalf*16 -------
    const int nn    = tid & 63;
    const int khalf = tid >> 6;
    const int ns = n0 + nn;
    const int b2 = ns / HW;
    const int r2 = ns - b2 * HW;
    const int h2 = r2 / Wd;
    const int w2 = r2 - h2 * Wd;

    const bool fast = (Cin & 31) == 0;
    // fast-path incremental (tap, channel) state; advances by 32 per stage
    int rF  = 0;
    int ciF = khalf * 16;

    auto stageTile = [&](int kt, _Float16* lbuf) {
        _Float16* ldst = &lbuf[nn * 32 + khalf * 16];
        if (fast) {
            int kh = rF / 3;
            int kw = rF - kh * 3;
            int ih = h2 + kh - 1;
            int iw = w2 + kw - 1;
            if (ih >= 0 && ih < H && iw >= 0 && iw < Wd) {
                const _Float16* gsrc =
                    X + ((size_t)(b2 * H + ih) * Wd + iw) * Cin + ciF;
                // LDS aperture keeps the byte offset in addr[31:0]
                unsigned lo = (unsigned)(uintptr_t)ldst;
                unsigned long long ga = (unsigned long long)(uintptr_t)gsrc;
                // async copy 32B: INST_OFFSET applies to both LDS and global
                asm volatile("global_load_async_to_lds_b128 %0, %1, off"
                             :: "v"(lo), "v"(ga) : "memory");
                asm volatile("global_load_async_to_lds_b128 %0, %1, off offset:16"
                             :: "v"(lo), "v"(ga) : "memory");
            } else {
                v16h z = {};
                *(v16h*)ldst = z;        // zero padding (halo) lanes
            }
            ciF += 32;
            if (ciF >= Cin) { ciF -= Cin; rF += 1; }
        } else {
            v16h vv;
#pragma unroll
            for (int j = 0; j < 16; ++j) {
                int k = kt + khalf * 16 + j;
                _Float16 v = (_Float16)0.0f;
                if (k < Ktot) {
                    int r  = k / Cin;
                    int ci = k - r * Cin;
                    int kh = r / 3;
                    int kw = r - kh * 3;
                    int ih = h2 + kh - 1;
                    int iw = w2 + kw - 1;
                    if (ih >= 0 && ih < H && iw >= 0 && iw < Wd)
                        v = X[((size_t)(b2 * H + ih) * Wd + iw) * Cin + ci];
                }
                vv[j] = v;
            }
            *(v16h*)ldst = vv;
        }
    };

    v8f acc0 = {}, acc1 = {}, acc2 = {}, acc3 = {};

    stageTile(0, ldsB[0]);
    asm volatile("s_wait_asynccnt 0x0" ::: "memory");
    __syncthreads();

    int p = 0;
    for (int k0 = 0; k0 < Ktot; k0 += 32) {
        // ------ A fragment: two contiguous 16B loads from packed weights ---
        const _Float16* pA = Wp + (size_t)m * kpad + k0;
        if (k0 + 32 < Ktot) __builtin_prefetch(pA + 32, 0, 1);
        v8h lo8 = *(const v8h*)(pA + kbaseA);
        v8h hi8 = *(const v8h*)(pA + kbaseA + 8);
        v16h a = __builtin_shufflevector(lo8, hi8, 0, 1, 2, 3, 4, 5, 6, 7,
                                                   8, 9, 10, 11, 12, 13, 14, 15);

        // ------ B fragments: contiguous 32B LDS reads ----------------------
        const _Float16* lb = ldsB[p];
        v16h bf0 = *(const v16h*)(&lb[(0 * 16 + bN) * 32 + kbaseB]);
        v16h bf1 = *(const v16h*)(&lb[(1 * 16 + bN) * 32 + kbaseB]);
        v16h bf2 = *(const v16h*)(&lb[(2 * 16 + bN) * 32 + kbaseB]);
        v16h bf3 = *(const v16h*)(&lb[(3 * 16 + bN) * 32 + kbaseB]);

        // ------ async-stage next tile into the other buffer ----------------
        if (k0 + 32 < Ktot) stageTile(k0 + 32, ldsB[p ^ 1]);

        acc0 = __builtin_amdgcn_wmma_f32_16x16x32_f16(false, a, false, bf0,
                                                      (short)0, acc0, false, false);
        acc1 = __builtin_amdgcn_wmma_f32_16x16x32_f16(false, a, false, bf1,
                                                      (short)0, acc1, false, false);
        acc2 = __builtin_amdgcn_wmma_f32_16x16x32_f16(false, a, false, bf2,
                                                      (short)0, acc2, false, false);
        acc3 = __builtin_amdgcn_wmma_f32_16x16x32_f16(false, a, false, bf3,
                                                      (short)0, acc3, false, false);
        asm volatile("s_wait_asynccnt 0x0" ::: "memory");
        __syncthreads();
        p ^= 1;
    }

    // ------ write D: NHWC => 8 channels contiguous, one v8h store each -----
    const int mo = (lane < 16) ? 0 : 8;
    const int coBase = ((blockIdx.y * 4 + wave) << 4) + mo;
    v8f bv = *(const v8f*)(Bias + coBase);
#pragma unroll
    for (int s = 0; s < 4; ++s) {
        int n    = n0 + s * 16 + bN;
        int bimg = n / HW;
        int rems = n - bimg * HW;
        int oh   = rems / Wd;
        int ow   = rems - oh * Wd;
        v8f av = (s == 0) ? acc0 : (s == 1) ? acc1 : (s == 2) ? acc2 : acc3;
        v8h ov;
#pragma unroll
        for (int v = 0; v < 8; ++v) {
            float val = av[v] + bv[v];
            if (relu) val = fmaxf(val, 0.0f);
            ov[v] = (_Float16)val;
        }
        *(v8h*)(Y + ((size_t)(bimg * H + oh) * Wd + ow) * Cout + coBase) = ov;
    }
}

// ---------------------------------------------------------------------------
// 2x2 max pool, stride 2, f16 NHWC
// ---------------------------------------------------------------------------
__global__ void pool_kernel(const _Float16* __restrict__ x,
                            _Float16* __restrict__ y, int C, int H, int Wd) {
    int Ho = H >> 1, Wo = Wd >> 1;
    size_t total = (size_t)NB * Ho * Wo * C;
    size_t i = (size_t)blockIdx.x * blockDim.x + threadIdx.x;
    if (i >= total) return;
    int c = (int)(i % C);
    size_t t = i / C;
    int wo = (int)(t % Wo); t /= Wo;
    int ho = (int)(t % Ho);
    int b  = (int)(t / Ho);
    const _Float16* p = x + ((size_t)(b * H + 2 * ho) * Wd + 2 * wo) * C + c;
    size_t rs = (size_t)Wd * C;
    float m = fmaxf(fmaxf((float)p[0], (float)p[C]),
                    fmaxf((float)p[rs], (float)p[rs + C]));
    y[i] = (_Float16)m;
}

// ---------------------------------------------------------------------------
// sum |a - b| over final VGG features -> acc (LDS ds_add_f32 reduction)
// ---------------------------------------------------------------------------
__global__ void l1diff_kernel(const _Float16* __restrict__ a,
                              const _Float16* __restrict__ b,
                              float* __restrict__ acc, int n) {
    __shared__ float s;
    if (threadIdx.x == 0) s = 0.0f;
    __syncthreads();
    int i = blockIdx.x * blockDim.x + threadIdx.x;
    float v = 0.0f;
    if (i < n) v = fabsf((float)a[i] - (float)b[i]);
    atomicAdd(&s, v);
    __syncthreads();
    if (threadIdx.x == 0) atomicAdd(acc, s);
}

// ---------------------------------------------------------------------------
// one pass over pixels: MSE, lightness hinge, TV(gray), TV(orig), quant min
// acc[0]=sum_sq, acc[1]=sum_hinge, acc[2]=sum_tv_gray, acc[3]=sum_tv_orig,
// acc[4]=sum_quant
// ---------------------------------------------------------------------------
__global__ void pixel_losses_kernel(const float* __restrict__ gray,
                                    const float* __restrict__ orig,
                                    const float* __restrict__ rest,
                                    float* __restrict__ acc) {
    __shared__ float s[5];
    if (threadIdx.x < 5) s[threadIdx.x] = 0.0f;
    __syncthreads();

    size_t i = (size_t)blockIdx.x * blockDim.x + threadIdx.x;
    float v_sq = 0.f, v_h = 0.f, v_tg = 0.f, v_to = 0.f, v_q = 0.f;
    if (i < (size_t)NPIX) {
        size_t HW = (size_t)NH * NW;
        size_t b = i / HW;
        size_t p = i - b * HW;
        int h = (int)(p / NW);
        int w = (int)(p - (size_t)h * NW);

        const float* o0 = orig + (b * 3 + 0) * HW;
        const float* o1 = orig + (b * 3 + 1) * HW;
        const float* o2 = orig + (b * 3 + 2) * HW;
        const float* r0 = rest + (b * 3 + 0) * HW;
        const float* r1 = rest + (b * 3 + 1) * HW;
        const float* r2 = rest + (b * 3 + 2) * HW;
        const float* gb = gray + b * HW;

        float d0 = o0[p] - r0[p], d1 = o1[p] - r1[p], d2 = o2[p] - r2[p];
        v_sq = d0 * d0 + d1 * d1 + d2 * d2;

        float g = gb[p];
        float lum = 0.299f * o0[p] + 0.587f * o1[p] + 0.114f * o2[p];
        v_h = fmaxf(fabsf(g - lum) - THRESH, 0.0f);

        if (h < NH - 1) v_tg += fabsf(gb[p + NW] - g);
        if (w < NW - 1) v_tg += fabsf(gb[p + 1] - g);
        if (h < NH - 1)
            v_to += fabsf(o0[p + NW] - o0[p]) + fabsf(o1[p + NW] - o1[p]) +
                    fabsf(o2[p + NW] - o2[p]);
        if (w < NW - 1)
            v_to += fabsf(o0[p + 1] - o0[p]) + fabsf(o1[p + 1] - o1[p]) +
                    fabsf(o2[p + 1] - o2[p]);

        // min over 256 uniform levels lv_j = j/127.5 - 1, running-min init 1.0
        float idx = rintf((g + 1.0f) * 127.5f);
        idx = fminf(fmaxf(idx, 0.0f), 255.0f);
        v_q = fminf(fabsf(g - (idx / 127.5f - 1.0f)), 1.0f);
    }
    atomicAdd(&s[0], v_sq);
    atomicAdd(&s[1], v_h);
    atomicAdd(&s[2], v_tg);
    atomicAdd(&s[3], v_to);
    atomicAdd(&s[4], v_q);
    __syncthreads();
    if (threadIdx.x < 5) atomicAdd(&acc[threadIdx.x], s[threadIdx.x]);
}

// ---------------------------------------------------------------------------
// combine accumulators into the scalar loss
// ---------------------------------------------------------------------------
__global__ void finalize_kernel(const float* __restrict__ acc,
                                const int* __restrict__ stage_p,
                                float* __restrict__ out) {
    if (threadIdx.x != 0 || blockIdx.x != 0) return;
    float i_loss  = acc[0] / (float)(NB * 3 * NH * NW);
    float l_loss  = acc[1] / (float)NPIX;
    float ls_loss = fabsf((3.0f * acc[2] - acc[3]) / (float)NB);
    float c_loss  = acc[5] / (float)FEATN;
    float g_loss  = l_loss + 1e-7f * c_loss + 0.5f * ls_loss;
    float q_loss  = acc[4];
    int stage = *stage_p;
    out[0] = (stage == 1) ? (i_loss + g_loss)
                          : (i_loss + 0.5f * g_loss + 10.0f * q_loss);
}

// ---------------------------------------------------------------------------
extern "C" void kernel_launch(void* const* d_in, const int* in_sizes, int n_in,
                              void* d_out, int out_size, void* d_ws, size_t ws_size,
                              hipStream_t stream) {
    (void)in_sizes; (void)n_in; (void)out_size; (void)ws_size;
    const float* gray  = (const float*)d_in[0];
    const float* orig  = (const float*)d_in[1];
    const float* rest  = (const float*)d_in[2];
    const int*   stage = (const int*)d_in[3];
    const float* Wv[11];
    const float* Bv[11];
    for (int k = 0; k < 11; ++k) {
        Wv[k] = (const float*)d_in[4 + k];
        Bv[k] = (const float*)d_in[15 + k];
    }

    static const int CH[11] = {64, 64, 128, 128, 256, 256, 256, 256, 512, 512, 512};

    // workspace layout:
    //   [0,256)         : float accumulators
    //   bufA, bufB      : 64MB each (f16 NHWC ping-pong activations)
    //   stash           : 8MB (VGG(g3) final features, f16)
    //   Wp[k]           : packed f16 weights per layer (K padded to mult 32)
    char* base = (char*)d_ws;
    float*    acc   = (float*)base;
    const size_t BUFB = 67108864ULL;                 // 8*256*256*64 * 2B
    _Float16* bufA  = (_Float16*)(base + 256);
    _Float16* bufB  = (_Float16*)(base + 256 + BUFB);
    _Float16* stash = (_Float16*)(base + 256 + 2 * BUFB);

    _Float16* Wp[11];
    int kpad[11];
    {
        size_t off = 256 + 2 * BUFB + (size_t)FEATN * sizeof(_Float16);
        int cin = 3;
        for (int k = 0; k < 11; ++k) {
            int kt = cin * 9;
            kpad[k] = ((kt + 31) / 32) * 32;
            Wp[k] = (_Float16*)(base + off);
            off += (size_t)CH[k] * kpad[k] * sizeof(_Float16);
            off = (off + 255) & ~(size_t)255;
            cin = CH[k];
        }
    }

    zero_acc_kernel<<<1, 32, 0, stream>>>(acc);

    // pack all layer weights to f16 GEMM layout (reused by both VGG passes)
    {
        int cin = 3;
        for (int k = 0; k < 11; ++k) {
            size_t tot = (size_t)CH[k] * kpad[k];
            pack_weights_kernel<<<(unsigned)((tot + 255) / 256), 256, 0, stream>>>(
                Wv[k], Wp[k], CH[k], cin, kpad[k]);
            cin = CH[k];
        }
    }

    for (int pass = 0; pass < 2; ++pass) {
        size_t packN = (size_t)NB * NH * NW * 3;
        pack_input_kernel<<<(unsigned)((packN + 255) / 256), 256, 0, stream>>>(
            pass == 0 ? gray : orig, bufA, pass == 0 ? 1 : 3);

        _Float16* cur = bufA;
        _Float16* nxt = bufB;
        int Hc = NH, Wc = NW, Cin = 3;
        for (int k = 0; k < 11; ++k) {
            int Cout = CH[k];
            int sp = NB * Hc * Wc;
            dim3 g((unsigned)(sp / 64), (unsigned)(Cout / 64));
            conv_wmma_kernel<<<g, 128, 0, stream>>>(cur, Wp[k], Bv[k], nxt,
                                                    Cin, Cout, Hc, Wc,
                                                    (k < 10) ? 1 : 0, kpad[k]);
            { _Float16* t = cur; cur = nxt; nxt = t; }
            if (k == 1 || k == 3 || k == 7) {
                size_t tot = (size_t)NB * Cout * (Hc / 2) * (Wc / 2);
                pool_kernel<<<(unsigned)((tot + 255) / 256), 256, 0, stream>>>(
                    cur, nxt, Cout, Hc, Wc);
                { _Float16* t = cur; cur = nxt; nxt = t; }
                Hc >>= 1; Wc >>= 1;
            }
            Cin = Cout;
        }
        if (pass == 0) {
            hipMemcpyAsync(stash, cur, (size_t)FEATN * sizeof(_Float16),
                           hipMemcpyDeviceToDevice, stream);
        } else {
            l1diff_kernel<<<(FEATN + 255) / 256, 256, 0, stream>>>(
                cur, stash, acc + 5, FEATN);
        }
    }

    pixel_losses_kernel<<<(NPIX + 255) / 256, 256, 0, stream>>>(gray, orig, rest, acc);
    finalize_kernel<<<1, 32, 0, stream>>>(acc, stage, (float*)d_out);
}
